// NodeFeatPredict_29635274342697
// MI455X (gfx1250) — matrix operand order repaired and checked by reading, not compiled, not run on hardware
//
#include <hip/hip_runtime.h>
#include <hip/hip_bf16.h>

// ---------------------------------------------------------------------------
// GCN forward (5 x GCNConv + ReLU + column softmax) for MI455X / gfx1250.
// Dense h@W done with v_wmma_f32_16x16x32_bf16 (bf16 in, f32 accumulate).
// Sparse symmetric-normalized aggregation done with native f32 atomics.
// ---------------------------------------------------------------------------

#define N_NODES 100000
#define N_EDGES 1600000
#define F_IN    256
#define HID     256
#define N_CLS   16

typedef __attribute__((ext_vector_type(16))) __bf16 v16bf;
typedef __attribute__((ext_vector_type(8)))  __bf16 v8bf;
typedef __attribute__((ext_vector_type(8)))  float  v8f;

// ---------------------------------------------------------------- degree ----
__global__ void k_deg_init(float* deg, int n) {
    int i = blockIdx.x * blockDim.x + threadIdx.x;
    if (i < n) deg[i] = 1.0f;                 // self loop contributes 1
}

__global__ void k_deg_accum(const int* __restrict__ dst, float* deg, int e) {
    int i = blockIdx.x * blockDim.x + threadIdx.x;
    if (i < e) unsafeAtomicAdd(&deg[dst[i]], 1.0f);
}

__global__ void k_deg_to_dis(float* dis, int n) {  // in place: deg -> rsqrt(deg)
    int i = blockIdx.x * blockDim.x + threadIdx.x;
    if (i < n) dis[i] = rsqrtf(dis[i]);       // deg >= 1 always
}

__global__ void k_edge_norm(const int* __restrict__ src, const int* __restrict__ dst,
                            const float* __restrict__ dis, float* __restrict__ enorm, int e) {
    int i = blockIdx.x * blockDim.x + threadIdx.x;
    if (i < e) enorm[i] = dis[src[i]] * dis[dst[i]];
}

// ------------------------------------------------------------ weight pack ---
// Pack W[K][ncols] (f32, K=256) into per-wave contiguous BF16 B-fragments:
// idx = ((ct*8 + ks)*32 + lane)*16 + e  with  K = ks*32 + (e>>3)*16 + (lane>>4)*8 + (e&7),
// col = ct*16 + (lane&15)   (matches 16-bit B 32x16 WMMA lane layout).
__global__ void k_pack_b(const float* __restrict__ W, __bf16* __restrict__ pack, int ncols) {
    int idx = blockIdx.x * blockDim.x + threadIdx.x;
    int total = (ncols >> 4) << 12;           // (ncols/16)*4096
    if (idx >= total) return;
    int e    = idx & 15;
    int lane = (idx >> 4) & 31;
    int ks   = (idx >> 9) & 7;
    int ct   = idx >> 12;
    int K    = ks * 32 + (e >> 3) * 16 + (lane >> 4) * 8 + (e & 7);
    int col  = ct * 16 + (lane & 15);
    pack[idx] = (__bf16)W[(size_t)K * ncols + col];
}

// ------------------------------------------------------------- f32 -> bf16 --
__global__ void k_cvt_bf16(const float* __restrict__ in, __bf16* __restrict__ out,
                           int n4, int relu) {
    int i = blockIdx.x * blockDim.x + threadIdx.x;
    if (i >= n4) return;
    float4 v = ((const float4*)in)[i];
    if (relu) {
        v.x = fmaxf(v.x, 0.0f); v.y = fmaxf(v.y, 0.0f);
        v.z = fmaxf(v.z, 0.0f); v.w = fmaxf(v.w, 0.0f);
    }
    __bf16* p = out + (size_t)i * 4;
    p[0] = (__bf16)v.x; p[1] = (__bf16)v.y; p[2] = (__bf16)v.z; p[3] = (__bf16)v.w;
}

// -------------------------------------------------------------- WMMA GEMM ---
// C[M][ncols] = A[M][256] (bf16) @ Wpack (bf16), f32 accumulate.
// One wave -> one 16x16 C tile; 8 k-steps of v_wmma_f32_16x16x32_bf16.
__global__ void k_gemm_wmma(const __bf16* __restrict__ Abf,
                            const __bf16* __restrict__ Bpack,
                            float* __restrict__ C, int M, int ncols) {
    const int lane = threadIdx.x & 31;
    const int wave = threadIdx.x >> 5;
    const int m0   = blockIdx.x * 16;
    const int nColTiles = ncols >> 4;
    const int ct   = blockIdx.y * (blockDim.x >> 5) + wave;
    if (ct >= nColTiles || m0 >= M) return;   // wave-uniform: EXEC stays all-1s

    const int half = lane >> 4;               // K half-split within the wave
    const int sub  = lane & 15;               // A row within tile / C column

    const __bf16* arow  = Abf + (size_t)(m0 + sub) * 256;
    const __bf16* bbase = Bpack + (((size_t)ct * 8) * 32 + lane) * 16;

    v8f c = {};
#pragma unroll
    for (int ks = 0; ks < 8; ++ks) {
        // A 16x32 bf16 fragment: lanes 0-15 hold K 0-7 & 16-23, lanes 16-31
        // hold K 8-15 & 24-31 (two contiguous 16B chunks per lane).
        v8bf alo = *(const v8bf*)(arow + ks * 32 + half * 8);
        v8bf ahi = *(const v8bf*)(arow + ks * 32 + 16 + half * 8);
        v16bf a = __builtin_shufflevector(alo, ahi,
                    0, 1, 2, 3, 4, 5, 6, 7, 8, 9, 10, 11, 12, 13, 14, 15);
        // B 32x16 bf16 fragment: pre-packed contiguous 32B per lane.
        v16bf b = *(const v16bf*)(bbase + (size_t)ks * 512);
        c = __builtin_amdgcn_wmma_f32_16x16x32_bf16(
                false, a, false, b, (short)0, c, false, false);
    }

    // C/D 16x16 f32 layout: VGPR r -> row m0 + 8*half + r, col ct*16 + sub.
    float* crow = C + (size_t)(m0 + half * 8) * ncols + ct * 16 + sub;
#pragma unroll
    for (int r = 0; r < 8; ++r) crow[(size_t)r * ncols] = c[r];
}

// ------------------------------------------------------------ aggregation ---
// agg[i][f] = dis[i]^2 * h[i][f] + bias[f]      (self-loop term + bias)
__global__ void k_init_agg(const float* __restrict__ h, const float* __restrict__ dis,
                           const float* __restrict__ bias, float* __restrict__ agg,
                           int n, int F) {
    int i = blockIdx.x * blockDim.x + threadIdx.x;
    if (i >= n * F) return;                   // n*F <= 25.6M, fits int
    int node = i / F, f = i - node * F;
    float d = dis[node];
    agg[i] = d * d * h[i] + bias[f];
}

// agg[dst] += norm[e] * h[src]   (4 floats per thread, native f32 atomics)
__global__ void k_scatter(const int* __restrict__ src, const int* __restrict__ dst,
                          const float* __restrict__ enorm, const float* __restrict__ h,
                          float* __restrict__ agg, int E, int F) {
    int cpe = F >> 2;                         // float4 chunks per edge
    int epb = blockDim.x / cpe;               // edges per block
    int e   = blockIdx.x * epb + threadIdx.x / cpe;
    if (e >= E) return;
    int f0  = (threadIdx.x % cpe) << 2;
    int s = src[e], d = dst[e];
    float w = enorm[e];
    float4 v = *(const float4*)(h + (size_t)s * F + f0);
    float* p = agg + (size_t)d * F + f0;
    unsafeAtomicAdd(p + 0, w * v.x);
    unsafeAtomicAdd(p + 1, w * v.y);
    unsafeAtomicAdd(p + 2, w * v.z);
    unsafeAtomicAdd(p + 3, w * v.w);
}

// -------------------------------------------------- softmax over dim 0 ------
__global__ void k_softmax_col(const float* __restrict__ h, float* __restrict__ out,
                              int n, int C) {
    const int c = blockIdx.x;                 // one block per class column
    __shared__ float red[256];
    float m = -INFINITY;
    for (int i = threadIdx.x; i < n; i += blockDim.x)
        m = fmaxf(m, h[(size_t)i * C + c]);
    red[threadIdx.x] = m; __syncthreads();
    for (int s = blockDim.x >> 1; s > 0; s >>= 1) {
        if (threadIdx.x < s) red[threadIdx.x] = fmaxf(red[threadIdx.x], red[threadIdx.x + s]);
        __syncthreads();
    }
    m = red[0]; __syncthreads();
    float sum = 0.0f;
    for (int i = threadIdx.x; i < n; i += blockDim.x)
        sum += __expf(h[(size_t)i * C + c] - m);
    red[threadIdx.x] = sum; __syncthreads();
    for (int s = blockDim.x >> 1; s > 0; s >>= 1) {
        if (threadIdx.x < s) red[threadIdx.x] += red[threadIdx.x + s];
        __syncthreads();
    }
    float inv = 1.0f / red[0];
    for (int i = threadIdx.x; i < n; i += blockDim.x)
        out[(size_t)i * C + c] = __expf(h[(size_t)i * C + c] - m) * inv;
}

// ---------------------------------------------------------------------------
extern "C" void kernel_launch(void* const* d_in, const int* in_sizes, int n_in,
                              void* d_out, int out_size, void* d_ws, size_t ws_size,
                              hipStream_t stream) {
    (void)in_sizes; (void)n_in; (void)out_size; (void)ws_size;

    const float* x   = (const float*)d_in[0];
    const int*   ei  = (const int*)d_in[1];      // [2, E]: row0 = src, row1 = dst
    const float* W1  = (const float*)d_in[2];
    const float* b1  = (const float*)d_in[3];
    const float* Wi  = (const float*)d_in[4];
    const float* bi  = (const float*)d_in[5];
    const float* W2  = (const float*)d_in[6];
    const float* b2  = (const float*)d_in[7];
    float*       out = (float*)d_out;

    const int* src = ei;
    const int* dst = ei + N_EDGES;

    // ---- workspace carve-up (all offsets 256B aligned) ----
    char* ws = (char*)d_ws;
    float*  dis    = (float*) (ws + 0);                  //   400,000 B
    float*  enorm  = (float*) (ws + 400128);             // 6,400,000 B
    __bf16* packW1 = (__bf16*)(ws + 6800128);            //   131,072 B
    __bf16* packWi = (__bf16*)(ws + 6931200);            //   131,072 B
    __bf16* packW2 = (__bf16*)(ws + 7062272);            //     8,192 B
    __bf16* hbf    = (__bf16*)(ws + 7070464);            // 51,200,000 B
    float*  tmp    = (float*) (ws + 58270464);           // 102,400,000 B
    float*  bufA   = (float*) (ws + 160670464);          // 102,400,000 B
    // total ~263 MB

    const int T = 256;

    // ---- graph normalization (computed once, reused by all 5 convs) ----
    k_deg_init  <<<(N_NODES + T - 1) / T, T, 0, stream>>>(dis, N_NODES);
    k_deg_accum <<<(N_EDGES + T - 1) / T, T, 0, stream>>>(dst, dis, N_EDGES);
    k_deg_to_dis<<<(N_NODES + T - 1) / T, T, 0, stream>>>(dis, N_NODES);
    k_edge_norm <<<(N_EDGES + T - 1) / T, T, 0, stream>>>(src, dst, dis, enorm, N_EDGES);

    // ---- pre-pack weights into WMMA B-fragment layout ----
    k_pack_b<<<(65536 + T - 1) / T, T, 0, stream>>>(W1, packW1, HID);
    k_pack_b<<<(65536 + T - 1) / T, T, 0, stream>>>(Wi, packWi, HID);
    k_pack_b<<<( 4096 + T - 1) / T, T, 0, stream>>>(W2, packW2, N_CLS);

    const int n4       = N_NODES * HID / 4;              // cvt thread count
    const int rowTiles = N_NODES / 16;                   // 6250 (exact)

    // ---- layers 1..4: 256 -> 256 GCN convs ----
    for (int layer = 0; layer < 4; ++layer) {
        const float*  hin   = (layer == 0) ? x      : bufA;
        const __bf16* packW = (layer == 0) ? packW1 : packWi;
        const float*  bias  = (layer == 0) ? b1     : bi;

        k_cvt_bf16<<<(n4 + T - 1) / T, T, 0, stream>>>(hin, hbf, n4, 0);
        k_gemm_wmma<<<dim3(rowTiles, 2), T, 0, stream>>>(hbf, packW, tmp, N_NODES, HID);
        k_init_agg<<<(N_NODES * HID + T - 1) / T, T, 0, stream>>>(tmp, dis, bias, bufA,
                                                                  N_NODES, HID);
        // 64 chunks/edge -> 4 edges per 256-thread block
        k_scatter<<<N_EDGES / 4, T, 0, stream>>>(src, dst, enorm, tmp, bufA,
                                                 N_EDGES, HID);
    }

    // ---- layer 5: ReLU -> 256 x 16 conv ----
    k_cvt_bf16<<<(n4 + T - 1) / T, T, 0, stream>>>(bufA, hbf, n4, 1);       // fused ReLU
    k_gemm_wmma<<<dim3(rowTiles, 1), 32, 0, stream>>>(hbf, packW2, tmp, N_NODES, N_CLS);
    k_init_agg<<<(N_NODES * N_CLS + T - 1) / T, T, 0, stream>>>(tmp, dis, b2, bufA,
                                                                N_NODES, N_CLS);
    // 4 chunks/edge -> 64 edges per 256-thread block
    k_scatter<<<N_EDGES / 64, T, 0, stream>>>(src, dst, enorm, tmp, bufA,
                                              N_EDGES, N_CLS);

    // ---- softmax over nodes (dim 0), one block per class ----
    k_softmax_col<<<N_CLS, T, 0, stream>>>(bufA, out, N_NODES, N_CLS);
}
